// PolarQuant_36481452212695
// MI455X (gfx1250) — matrix operand order repaired and checked by reading, not compiled, not run on hardware
//
#include <hip/hip_runtime.h>

typedef __attribute__((ext_vector_type(16))) _Float16 v16h;
typedef __attribute__((ext_vector_type(8)))  _Float16 v8h;
typedef __attribute__((ext_vector_type(4)))  _Float16 v4h;
typedef __attribute__((ext_vector_type(8)))  float    v8f;
typedef __attribute__((ext_vector_type(4)))  float    v4f;

constexpr int kD     = 128;
constexpr int kBatch = 262144;
constexpr int kHPad  = 136;            // padded LDS row stride (f16 elems) -> conflict-free b128
constexpr int kTiles = kBatch / 16;    // 16384 tiles of 16 rows

// Y(16x128) += X(16x128) * H(128x128), f16 WMMA, f32 accum.
// A layout (16-bit 16x32): lane m=lane&15, K sub-offset 8*(lane>>4); regs 0..3 = 8 contig K, 4..7 = +16.
// B layout (32x16):        lane n=lane&15, K sub-offset 8*(lane>>4); regs 0..3 = 8 contig K, 4..7 = +16.
__device__ __forceinline__ void mm16x128(const _Float16* myX, const _Float16* Hc,
                                         int m_a, int k_ah, int n_b, int k_bh,
                                         v8f acc[8]) {
#pragma unroll
  for (int k0 = 0; k0 < 128; k0 += 32) {
    v8h alo = *(const v8h*)(myX + m_a * kHPad + k0 + k_ah);
    v8h ahi = *(const v8h*)(myX + m_a * kHPad + k0 + 16 + k_ah);
    v16h a = __builtin_shufflevector(alo, ahi, 0,1,2,3,4,5,6,7,8,9,10,11,12,13,14,15);
#pragma unroll
    for (int t = 0; t < 8; ++t) {
      const _Float16* hp = Hc + (t * 16 + n_b) * kHPad + k0 + k_bh; // H symmetric: Hc[n][k]==H[k][n]
      v8h blo = *(const v8h*)hp;
      v8h bhi = *(const v8h*)(hp + 16);
      v16h b = __builtin_shufflevector(blo, bhi, 0,1,2,3,4,5,6,7,8,9,10,11,12,13,14,15);
      acc[t] = __builtin_amdgcn_wmma_f32_16x16x32_f16(false, a, false, b,
                                                      (short)0, acc[t], false, false);
    }
  }
}

__global__ __launch_bounds__(128) void polarquant_wmma(
    const float* __restrict__ x, const float* __restrict__ signs,
    const float* __restrict__ cent, float* __restrict__ xhat_out,
    int* __restrict__ idx_out)
{
  __shared__ _Float16 Hc[128 * kHPad];     // 34816 B  Hadamard table
  __shared__ _Float16 Xt[4][16 * kHPad];   // 17408 B  per-wave staging tile
  __shared__ v4f      QT[256];             //  4096 B  quantizer LUT: {bnd, c_lo, c_hi, idx0}

  const int lane = threadIdx.x & 31;
  const int wave = threadIdx.x >> 5;

  // Build H_128 (Sylvester, entries +-1) once per workgroup.
  for (int i = threadIdx.x; i < 128 * 128; i += 128) {
    int n = i >> 7, k = i & 127;
    Hc[n * kHPad + k] = (_Float16)((__builtin_popcount(n & k) & 1) ? -1.0f : 1.0f);
  }

  // Quantizer constants (uniform loads -> SGPRs). Boundaries pre-scaled by sqrt(D)
  // so the raw WMMA accumulator (y * sqrt(D), ~N(0,1)) is compared directly.
  const float sqrtD = 11.313708498984761f;
  float c[16], bndS[15];
#pragma unroll
  for (int i = 0; i < 16; ++i) c[i] = cent[i];
#pragma unroll
  for (int i = 0; i < 15; ++i) bndS[i] = 0.5f * (c[i] + c[i + 1]) * sqrtD;

  // LUT over y_raw in [-2, 2), 256 cells of width 1/64 (< min boundary gap ~0.157).
  // Entry j: idx0 = #{bndS < leftedge}, b = bndS[idx0] (or +inf), c_lo=c[idx0], c_hi=c[idx0+1].
  for (int j = threadIdx.x; j < 256; j += 128) {
    float e = (float)(j - 128) * (1.0f / 64.0f);
    int i0 = 0;
    float b = __builtin_inff();
    float clo = c[0], chi = c[1];
#pragma unroll
    for (int i = 0; i < 15; ++i) i0 += (bndS[i] < e) ? 1 : 0;
#pragma unroll
    for (int i = 14; i >= 0; --i) if (!(bndS[i] < e)) b = bndS[i];      // first boundary >= e
#pragma unroll
    for (int i = 0; i < 15; ++i) if (bndS[i] < e) clo = c[i + 1];       // c[i0]
#pragma unroll
    for (int i = 0; i < 15; ++i) if (bndS[i] < e) chi = c[(i + 2) < 15 ? (i + 2) : 15]; // c[min(i0+1,15)]
    v4f q = { b, clo, chi, __int_as_float(i0) };
    QT[j] = q;
  }
  __syncthreads();

  const int m_a  = lane & 15;          // A row
  const int k_ah = (lane >> 4) * 8;    // A K sub-offset
  const int n_b  = lane & 15;          // B col within 16-tile
  const int k_bh = (lane >> 4) * 8;    // B K sub-offset
  const int m_c  = (lane >> 4) * 8;    // C/D row offset (reg r -> M = m_c + r)
  const float scale = 0.08838834764831845f;  // 1/sqrt(128)

  v4f sg4 = *(const v4f*)(signs + 4 * lane);   // signs for staging cols 4*lane..+3
  float sc_n[8];                               // scale * signs for output cols
#pragma unroll
  for (int t = 0; t < 8; ++t) sc_n[t] = signs[t * 16 + n_b] * scale;

  _Float16* myX = &Xt[wave][0];
  const v8f zero8 = {0.f,0.f,0.f,0.f,0.f,0.f,0.f,0.f};
  const int nWaves = (int)gridDim.x * 4;

  for (int tile = (int)blockIdx.x * 4 + wave; tile < kTiles; tile += nWaves) {
    const float* xt = x + (size_t)tile * 16 * kD;

    // Stage tile: f32 global -> (x * signs) as f16 in LDS. One row per b128 load across the wave.
#pragma unroll
    for (int r = 0; r < 16; ++r) {
      v4f v = *(const v4f*)(xt + r * kD + 4 * lane);
      v *= sg4;
      v4h h = { (_Float16)v.x, (_Float16)v.y, (_Float16)v.z, (_Float16)v.w };
      *(v4h*)(myX + r * kHPad + 4 * lane) = h;
    }
    asm volatile("s_wait_dscnt 0" ::: "memory");

    // Pass 1: Y_raw = (X .* signs) * H   (reference y = Y_raw / sqrt(D))
    v8f acc[8];
#pragma unroll
    for (int t = 0; t < 8; ++t) acc[t] = zero8;
    mm16x128(myX, Hc, m_a, k_ah, n_b, k_bh, acc);

    // Quantize via packed LUT + one exact correction compare; emit indices; restage y_hat.
#pragma unroll
    for (int t = 0; t < 8; ++t) {
#pragma unroll
      for (int r = 0; r < 8; ++r) {
        float y = acc[t][r];
        float u = __builtin_fmaf(y, 64.0f, 128.0f);
        u = fminf(fmaxf(u, 0.0f), 255.0f);
        int j = (int)u;
        v4f q = QT[j];
        bool g = y > q.x;
        int idx = __float_as_int(q.w) + (g ? 1 : 0);
        float yh = g ? q.z : q.y;               // centroid in reference space
        int row = m_c + r, col = t * 16 + n_b;
        idx_out[(size_t)(tile * 16 + row) * kD + col] = idx;
        myX[row * kHPad + col] = (_Float16)yh;
      }
    }
    asm volatile("s_wait_dscnt 0" ::: "memory");

    // Pass 2: Xhat_raw = Yhat * H
#pragma unroll
    for (int t = 0; t < 8; ++t) acc[t] = zero8;
    mm16x128(myX, Hc, m_a, k_ah, n_b, k_bh, acc);

    // Emit x_hat = Xhat_raw / sqrt(D) .* signs (pre-merged per-column factor)
#pragma unroll
    for (int t = 0; t < 8; ++t) {
#pragma unroll
      for (int r = 0; r < 8; ++r) {
        int row = m_c + r, col = t * 16 + n_b;
        xhat_out[(size_t)(tile * 16 + row) * kD + col] = acc[t][r] * sc_n[t];
      }
    }
  }
}

extern "C" void kernel_launch(void* const* d_in, const int* in_sizes, int n_in,
                              void* d_out, int out_size, void* d_ws, size_t ws_size,
                              hipStream_t stream) {
  const float* x     = (const float*)d_in[0];
  const float* signs = (const float*)d_in[1];
  const float* cent  = (const float*)d_in[2];
  float* xhat = (float*)d_out;
  int*   idx  = (int*)((float*)d_out + (size_t)kBatch * kD);
  hipLaunchKernelGGL(polarquant_wmma, dim3(1024), dim3(128), 0, stream,
                     x, signs, cent, xhat, idx);
}
